// CompressiveMemory_15272903704587
// MI455X (gfx1250) — matrix operand order repaired and checked by reading, compiled, not run on hardware
//
#include <hip/hip_runtime.h>
#include <cstdint>

#define DEV __device__ __forceinline__

typedef _Float16 h16;
typedef __attribute__((ext_vector_type(16))) _Float16 v16h;
typedef __attribute__((ext_vector_type(8)))  _Float16 v8h;
typedef __attribute__((ext_vector_type(8)))  float    v8f;

constexpr int B_ = 2, S_ = 8192, D_ = 2048, H_ = 16, DK_ = 128, DV_ = 128;
constexpr int SEG_ = 512, NSEG_ = 16;

// ----------------------------------------------------- async copy (CDNA5 path)
#if defined(__has_builtin)
#if __has_builtin(__builtin_amdgcn_global_load_async_to_lds_b128) && \
    __has_builtin(__builtin_amdgcn_s_wait_asynccnt)
#define HAVE_ASYNC_LDS 1
#endif
#endif

#if defined(HAVE_ASYNC_LDS)
typedef int v4i __attribute__((vector_size(16)));          // gcc-style int4
typedef __attribute__((address_space(1))) v4i* gptr4_t;    // global int4*
typedef __attribute__((address_space(3))) v4i* lptr4_t;    // LDS int4*
DEV void async_copy16(const h16* g, h16* l) {
  // 16B per lane, memory -> LDS, tracked by ASYNCcnt (no VGPR data staging)
  __builtin_amdgcn_global_load_async_to_lds_b128(
      (gptr4_t)(uintptr_t)g, (lptr4_t)(unsigned)(uintptr_t)l, 0, 0);
}
DEV void async_wait_group() { __builtin_amdgcn_s_wait_asynccnt(5); }
DEV void async_wait_all()   { __builtin_amdgcn_s_wait_asynccnt(0); }
#else
DEV void async_copy16(const h16* g, h16* l) { *(v8h*)l = *(const v8h*)g; }
DEV void async_wait_group() {}
DEV void async_wait_all()   {}
#endif

// ---------------------------------------------------------------- WMMA helpers
DEV v8f wmma16(v16h a, v16h b, v8f c) {
  return __builtin_amdgcn_wmma_f32_16x16x32_f16(
      false, a, false, b, (short)0, c, false, false);
}

// A-fragment: lane L holds row m=L&15; halves 0..15 -> k = 16*(L>>4)+0..15
// (contiguous order, matches B fragments)
DEV v16h load_a_frag(const h16* A, int lda, int lane) {
  int m = lane & 15, g = lane >> 4;
  const h16* p = A + (size_t)m * lda + g * 16;
  v8h lo = *(const v8h*)p;
  v8h hi = *(const v8h*)(p + 8);
  v16h r;
#pragma unroll
  for (int i = 0; i < 8; ++i) { r[i] = lo[i]; r[i + 8] = hi[i]; }
  return r;
}

// B-fragment from B^T (N x K row-major): lane L holds col n=L&15,
// k = 16*(L>>4) + h, h=0..15 -> one contiguous 32B run per lane.
DEV v16h load_b_frag(const h16* Bt, int ldb, int lane) {
  int n = lane & 15, g = lane >> 4;
  return *(const v16h*)(Bt + (size_t)n * ldb + g * 16);
}
// same, via two 16B loads (for padded LDS rows that are only 16B aligned)
DEV v16h load_b_frag2(const h16* Bt, int ldb, int lane) {
  int n = lane & 15, g = lane >> 4;
  const h16* p = Bt + (size_t)n * ldb + g * 16;
  v8h lo = *(const v8h*)p;
  v8h hi = *(const v8h*)(p + 8);
  v16h r;
#pragma unroll
  for (int i = 0; i < 8; ++i) { r[i] = lo[i]; r[i + 8] = hi[i]; }
  return r;
}

// ---------------------------------------------------------------- conversions
__global__ void cvt_f16_kernel(const float* __restrict__ in,
                               h16* __restrict__ out, size_t n) {
  size_t i = ((size_t)blockIdx.x * blockDim.x + threadIdx.x) * 4;
  if (i + 3 < n) {
    float4 v = *(const float4*)(in + i);
    out[i] = (h16)v.x; out[i + 1] = (h16)v.y;
    out[i + 2] = (h16)v.z; out[i + 3] = (h16)v.w;
  }
}

// Wt[n*K + k] = (f16) W[k*N + n]   (tiled through LDS)
__global__ void transpose_cvt_kernel(const float* __restrict__ W,
                                     h16* __restrict__ Wt, int K, int N) {
  __shared__ float tile[32][33];
  int nb = blockIdx.x * 32, kb = blockIdx.y * 32;
  int tx = threadIdx.x & 31, ty = threadIdx.x >> 5;
#pragma unroll
  for (int i = ty; i < 32; i += 8) tile[i][tx] = W[(size_t)(kb + i) * N + nb + tx];
  __syncthreads();
#pragma unroll
  for (int i = ty; i < 32; i += 8)
    Wt[(size_t)(nb + i) * K + kb + tx] = (h16)tile[tx][i];
}

// ---------------------------------------------------------------- GEMM
// C(MxN) = A(MxK f16) * B(KxN) given as Bt(NxK f16).
// Block 256 thr = 8 waves (2Mx4N); wave tile 32x64; block tile 64x256, BK=32.
// A/B tiles double-buffered in LDS, filled by async DMA (ASYNCcnt pipelined).
constexpr int BM = 64, BN = 256, BK = 32;
constexpr int LDS_K = BK + 8;  // 40-half row stride: conflict-free frag reads

template <bool HALF_OUT>
__global__ __launch_bounds__(256) void gemm_f16_kernel(
    const h16* __restrict__ A, const h16* __restrict__ Bt,
    void* __restrict__ Cout, int M, int N, int K) {
  __shared__ __align__(16) h16 As[2][BM][LDS_K];
  __shared__ __align__(16) h16 Bs[2][BN][LDS_K];

  const int tid = threadIdx.x;
  const int lane = tid & 31, wave = tid >> 5;
  const int mblk = blockIdx.y * BM, nblk = blockIdx.x * BN;
  const int wm = (wave >> 2) * 32, wn = (wave & 3) * 64;
  const int cr = tid >> 2, cc = (tid & 3) * 8;  // copy: row, 8-half chunk col

  // 5 x 16B async ops per thread per K-step: 1 A chunk + 4 B chunks
  auto issue = [&](int buf, int k0) {
    async_copy16(A + (size_t)(mblk + cr) * K + k0 + cc, &As[buf][cr][cc]);
    const h16* gb = Bt + (size_t)(nblk + cr) * K + k0 + cc;
#pragma unroll
    for (int j = 0; j < 4; ++j)
      async_copy16(gb + (size_t)(j * 64) * K, &Bs[buf][cr + 64 * j][cc]);
  };

  const int nsteps = K / BK;
  issue(0, 0);
  v8f acc[2][4] = {};
  for (int i = 0; i < nsteps; ++i) {
    const int buf = i & 1;
    if (i + 1 < nsteps) {
      issue(buf ^ 1, (i + 1) * BK);
      async_wait_group();   // <=5 outstanding: group i has landed in LDS
    } else {
      async_wait_all();
    }
    __syncthreads();

    v16h a0 = load_a_frag(&As[buf][wm][0], LDS_K, lane);
    v16h a1 = load_a_frag(&As[buf][wm + 16][0], LDS_K, lane);
    v16h b[4];
#pragma unroll
    for (int j = 0; j < 4; ++j)
      b[j] = load_b_frag2(&Bs[buf][wn + 16 * j][0], LDS_K, lane);
#pragma unroll
    for (int j = 0; j < 4; ++j) {
      acc[0][j] = wmma16(a0, b[j], acc[0][j]);
      acc[1][j] = wmma16(a1, b[j], acc[1][j]);
    }
    __syncthreads();  // all waves done reading buf before it is re-filled
  }

  int cm = (lane >> 4) * 8, cn = lane & 15;
#pragma unroll
  for (int i = 0; i < 2; ++i)
#pragma unroll
    for (int j = 0; j < 4; ++j)
#pragma unroll
      for (int r = 0; r < 8; ++r) {
        size_t off = (size_t)(mblk + wm + i * 16 + cm + r) * N
                   + nblk + wn + j * 16 + cn;
        if (HALF_OUT) ((h16*)Cout)[off] = (h16)acc[i][j][r];
        else          ((float*)Cout)[off] = acc[i][j][r];
      }
}

// ------------------------------------------ per-segment sigk^T @ v and ksum
__global__ __launch_bounds__(256) void seg_kv_stats_kernel(
    const h16* __restrict__ kh, const h16* __restrict__ vh,
    float* __restrict__ memcT, float* __restrict__ ksum) {
  int seg = blockIdx.x, bh = blockIdx.y, tid = threadIdx.x;
  const h16* kb = kh + ((size_t)bh * S_ + (size_t)seg * SEG_) * DK_;
  const h16* vb = vh + ((size_t)bh * S_ + (size_t)seg * SEG_) * DV_;
  __shared__ float sk[32][DK_];
  __shared__ float sv[32][DV_];
  int dv0 = (tid >> 4) * 8, dk0 = (tid & 15) * 8;
  float acc[8][8] = {};
  float ks = 0.f;
  for (int s0 = 0; s0 < SEG_; s0 += 32) {
    __syncthreads();
    for (int i = tid; i < 32 * 16; i += 256) {
      int s = i >> 4, c = (i & 15) * 8;
      v8h kk = *(const v8h*)(kb + (size_t)(s0 + s) * DK_ + c);
      v8h vv = *(const v8h*)(vb + (size_t)(s0 + s) * DV_ + c);
#pragma unroll
      for (int j = 0; j < 8; ++j) {
        float x = (float)kk[j];
        sk[s][c + j] = x > 0.f ? x + 1.f : __expf(x);
        sv[s][c + j] = (float)vv[j];
      }
    }
    __syncthreads();
#pragma unroll 4
    for (int s = 0; s < 32; ++s) {
      float vr[8], kr[8];
#pragma unroll
      for (int j = 0; j < 8; ++j) { vr[j] = sv[s][dv0 + j]; kr[j] = sk[s][dk0 + j]; }
#pragma unroll
      for (int i = 0; i < 8; ++i)
#pragma unroll
        for (int j = 0; j < 8; ++j) acc[i][j] += vr[i] * kr[j];
    }
    if (tid < DK_)
#pragma unroll 8
      for (int s = 0; s < 32; ++s) ks += sk[s][tid];
  }
  float* outp = memcT + ((size_t)seg * (B_ * H_) + bh) * DV_ * DK_;
#pragma unroll
  for (int i = 0; i < 8; ++i)
#pragma unroll
    for (int j = 0; j < 8; ++j)
      outp[(size_t)(dv0 + i) * DK_ + dk0 + j] = acc[i][j];
  if (tid < DK_) ksum[((size_t)seg * (B_ * H_) + bh) * DK_ + tid] = ks;
}

// exclusive prefix of memcT -> mempT (f16), inclusive prefix of ksum -> kz
__global__ __launch_bounds__(256) void seg_prefix_kernel(
    const float* __restrict__ memcT, const float* __restrict__ ksum,
    h16* __restrict__ mempT, float* __restrict__ kz) {
  int bh = blockIdx.x, tid = threadIdx.x;
  const size_t segstr = (size_t)(B_ * H_) * DV_ * DK_;
  for (int p = tid; p < DV_ * DK_; p += 256) {
    size_t base = (size_t)bh * DV_ * DK_ + p;
    float run = 0.f;
    for (int s = 0; s < NSEG_; ++s) {
      mempT[base + (size_t)s * segstr] = (h16)run;
      run += memcT[base + (size_t)s * segstr];
    }
  }
  if (tid < DK_) {
    size_t base = (size_t)bh * DK_ + tid;
    float run = 0.f;
    for (int s = 0; s < NSEG_; ++s) {
      run += ksum[base + (size_t)s * (B_ * H_ * DK_)];
      kz[base + (size_t)s * (B_ * H_ * DK_)] = run;
    }
  }
}

// ---------------------------------------------------------------- attention
// grid (8, NSEG, B*H); 128 thr = 4 waves; wave owns 16 q-rows.
__global__ __launch_bounds__(128) void seg_attn_kernel(
    const h16* __restrict__ qh, const h16* __restrict__ kh,
    const h16* __restrict__ vh, const h16* __restrict__ mempT,
    const float* __restrict__ kz, const float* __restrict__ betas,
    h16* __restrict__ attb) {
  __shared__ __align__(16) h16 vT[DV_][128 + 8];
  __shared__ __align__(16) h16 Psc[4][16][128 + 8];
  __shared__ float rsum[4][16];

  const int lane = threadIdx.x & 31, wave = threadIdx.x >> 5;
  const int qt = blockIdx.x, seg = blockIdx.y, bh = blockIdx.z;
  const int hh = bh & (H_ - 1), b = bh >> 4;
  const h16* qhd = qh + (size_t)bh * S_ * DK_;
  const h16* khd = kh + (size_t)bh * S_ * DK_;
  const h16* vhd = vh + (size_t)bh * S_ * DV_;
  const int mrow = seg * SEG_ + qt * 64 + wave * 16;

  v16h qa[4], sqa[4];
  float rs = 0.f;
#pragma unroll
  for (int kk = 0; kk < 4; ++kk) {
    qa[kk] = load_a_frag(qhd + (size_t)mrow * DK_ + kk * 32, DK_, lane);
#pragma unroll
    for (int t = 0; t < 16; ++t) {
      float x = (float)qa[kk][t];
      float s = x > 0.f ? x + 1.f : __expf(x);
      sqa[kk][t] = (h16)s;
      rs += s;
    }
  }
  rs += __shfl_xor(rs, 16, 32);
  if (lane < 16) rsum[wave][lane] = rs;

  const float scale = 0.08838834764831845f;  // 1/sqrt(128)
  float mmax[8], lsum[8];
  v8f oacc[8] = {};
#pragma unroll
  for (int r = 0; r < 8; ++r) { mmax[r] = -1e30f; lsum[r] = 0.f; }
  const int cm = lane >> 4, cn = lane & 15;

  for (int c = 0; c < 4; ++c) {
    __syncthreads();
    for (int i = threadIdx.x; i < 128 * 16; i += 128) {
      int s = i >> 4, dg = (i & 15) * 8;
      v8h vv = *(const v8h*)(vhd + (size_t)(seg * SEG_ + c * 128 + s) * DV_ + dg);
#pragma unroll
      for (int j = 0; j < 8; ++j) vT[dg + j][s] = vv[j];
    }
    __syncthreads();

    v8f sc[8] = {};
#pragma unroll
    for (int t = 0; t < 8; ++t) {
      int ncol = seg * SEG_ + c * 128 + t * 16;
#pragma unroll
      for (int kk = 0; kk < 4; ++kk) {
        v16h bf = load_b_frag(khd + (size_t)ncol * DK_ + kk * 32, DK_, lane);
        sc[t] = wmma16(qa[kk], bf, sc[t]);
      }
    }
#pragma unroll
    for (int r = 0; r < 8; ++r) {
      float mx = -1e30f;
#pragma unroll
      for (int t = 0; t < 8; ++t) { sc[t][r] *= scale; mx = fmaxf(mx, sc[t][r]); }
#pragma unroll
      for (int d = 1; d < 16; d <<= 1) mx = fmaxf(mx, __shfl_xor(mx, d, 32));
      float mn = fmaxf(mmax[r], mx);
      float sf = __expf(mmax[r] - mn);
      mmax[r] = mn;
      lsum[r] *= sf;
#pragma unroll
      for (int t = 0; t < 8; ++t) oacc[t][r] *= sf;
      float ps = 0.f;
#pragma unroll
      for (int t = 0; t < 8; ++t) { float p = __expf(sc[t][r] - mn); sc[t][r] = p; ps += p; }
#pragma unroll
      for (int d = 1; d < 16; d <<= 1) ps += __shfl_xor(ps, d, 32);
      lsum[r] += ps;
    }
#pragma unroll
    for (int t = 0; t < 8; ++t)
#pragma unroll
      for (int r = 0; r < 8; ++r)
        Psc[wave][r + 8 * cm][t * 16 + cn] = (h16)sc[t][r];
    __syncthreads();
#pragma unroll
    for (int kk = 0; kk < 4; ++kk) {
      v16h pa = load_a_frag(&Psc[wave][0][0] + kk * 32, 128 + 8, lane);
#pragma unroll
      for (int t = 0; t < 8; ++t) {
        v16h vb = load_b_frag2(&vT[t * 16][0] + kk * 32, 128 + 8, lane);
        oacc[t] = wmma16(pa, vb, oacc[t]);
      }
    }
  }

  v8f num[8] = {};
  const h16* mp = mempT + ((size_t)seg * (B_ * H_) + bh) * DV_ * DK_;
#pragma unroll
  for (int kk = 0; kk < 4; ++kk)
#pragma unroll
    for (int t = 0; t < 8; ++t) {
      v16h bf = load_b_frag(mp + (size_t)(t * 16) * DK_ + kk * 32, DK_, lane);
      num[t] = wmma16(sqa[kk], bf, num[t]);
    }

  const float* kzp = kz + ((size_t)seg * (B_ * H_) + bh) * DK_;
  h16* ob = attb + (((size_t)(b * NSEG_ + seg) * H_ + hh) * SEG_) * DV_;
#pragma unroll
  for (int t = 0; t < 8; ++t) {
    int dv = t * 16 + cn;
    float kzv = kzp[dv];
    float bet = 1.f / (1.f + __expf(-betas[hh * DV_ + dv]));
#pragma unroll
    for (int r = 0; r < 8; ++r) {
      int m = r + 8 * cm;
      float am = num[t][r] / (rsum[wave][m] * kzv);
      float ad = oacc[t][r] / lsum[r];
      ob[(size_t)(qt * 64 + wave * 16 + m) * DV_ + dv] = (h16)(bet * am + (1.f - bet) * ad);
    }
  }
}

// ---------------------------------------------------------------- launcher
extern "C" void kernel_launch(void* const* d_in, const int* in_sizes, int n_in,
                              void* d_out, int out_size, void* d_ws, size_t ws_size,
                              hipStream_t stream) {
  (void)in_sizes; (void)n_in; (void)out_size; (void)ws_size;
  const float* x  = (const float*)d_in[0];
  const float* Wq = (const float*)d_in[1];
  const float* Wk = (const float*)d_in[2];
  const float* Wv = (const float*)d_in[3];
  const float* betas = (const float*)d_in[4];
  const float* Wo = (const float*)d_in[5];

  char* w = (char*)d_ws;
  auto alloc = [&](size_t bytes) -> void* {
    void* p = (void*)w; w += (bytes + 255) & ~(size_t)255; return p;
  };
  const size_t NX = (size_t)B_ * S_ * D_;
  const size_t NW = (size_t)D_ * D_;
  h16* xh   = (h16*)alloc(NX * 2);
  h16* Wqt  = (h16*)alloc(NW * 2);
  h16* Wkt  = (h16*)alloc(NW * 2);
  h16* Wvt  = (h16*)alloc(NW * 2);
  h16* Wot  = (h16*)alloc(NW * 2);
  h16* qh   = (h16*)alloc(NX * 2);
  h16* kh   = (h16*)alloc(NX * 2);
  h16* vh   = (h16*)alloc(NX * 2);
  h16* attb = (h16*)alloc(NX * 2);
  float* memcT = (float*)alloc((size_t)NSEG_ * B_ * H_ * DV_ * DK_ * 4);
  h16*   mempT = (h16*)alloc((size_t)NSEG_ * B_ * H_ * DV_ * DK_ * 2);
  float* ksum  = (float*)alloc((size_t)NSEG_ * B_ * H_ * DK_ * 4);
  float* kz    = (float*)alloc((size_t)NSEG_ * B_ * H_ * DK_ * 4);

  cvt_f16_kernel<<<dim3((unsigned)(NX / 1024)), 256, 0, stream>>>(x, xh, NX);
  dim3 tg(D_ / 32, D_ / 32);
  transpose_cvt_kernel<<<tg, 256, 0, stream>>>(Wq, Wqt, D_, D_);
  transpose_cvt_kernel<<<tg, 256, 0, stream>>>(Wk, Wkt, D_, D_);
  transpose_cvt_kernel<<<tg, 256, 0, stream>>>(Wv, Wvt, D_, D_);
  transpose_cvt_kernel<<<tg, 256, 0, stream>>>(Wo, Wot, D_, D_);

  const int M = B_ * S_;
  dim3 gg(D_ / BN, M / BM);
  gemm_f16_kernel<true><<<gg, 256, 0, stream>>>(xh, Wqt, qh, M, D_, D_);
  gemm_f16_kernel<true><<<gg, 256, 0, stream>>>(xh, Wkt, kh, M, D_, D_);
  gemm_f16_kernel<true><<<gg, 256, 0, stream>>>(xh, Wvt, vh, M, D_, D_);

  seg_kv_stats_kernel<<<dim3(NSEG_, B_ * H_), 256, 0, stream>>>(kh, vh, memcT, ksum);
  seg_prefix_kernel<<<dim3(B_ * H_), 256, 0, stream>>>(memcT, ksum, mempT, kz);

  seg_attn_kernel<<<dim3(SEG_ / 64, NSEG_, B_ * H_), 128, 0, stream>>>(
      qh, kh, vh, mempT, kz, betas, attb);

  gemm_f16_kernel<false><<<gg, 256, 0, stream>>>(attb, Wot, (float*)d_out, M, D_, D_);
}
